// AttentionBlock_37460704756348
// MI455X (gfx1250) — matrix operand order repaired and checked by reading, not compiled, not run on hardware
//
#include <hip/hip_runtime.h>

// B=4, N=2048, C=1024, H=16, D=64. All fp32 in/out; internal GEMMs in f16 with
// fp32 accumulation via v_wmma_f32_16x16x32_f16 (the fast CDNA5 matrix path).
// Contiguous tile staging uses gfx1250 GLOBAL_LOAD_ASYNC_TO_LDS_B128 (ASYNCcnt)
// via inline asm; transposed staging stays on the register path.

typedef _Float16 v16h __attribute__((ext_vector_type(16)));
typedef _Float16 v8h  __attribute__((ext_vector_type(8)));
typedef _Float16 v4h  __attribute__((ext_vector_type(4)));
typedef float    v8f  __attribute__((ext_vector_type(8)));

__device__ __forceinline__ v8f wmma16(v16h a, v16h b, v8f c) {
  return __builtin_amdgcn_wmma_f32_16x16x32_f16(false, a, false, b, (short)0, c,
                                                false, false);
}

// Load a 16-element fragment as two contiguous 16B loads at element offsets
// o0 and o1 from p. Works for LDS and global pointers.
__device__ __forceinline__ v16h frag2(const _Float16* p, int o0, int o1) {
  v8h lo = *(const v8h*)(p + o0);
  v8h hi = *(const v8h*)(p + o1);
  v16h r;
#pragma unroll
  for (int i = 0; i < 8; i++) { r[i] = lo[i]; r[i + 8] = hi[i]; }
  return r;
}

// LDS byte offset of a __shared__ pointer (generic -> addrspace(3) -> u32).
__device__ __forceinline__ unsigned lds_off(const void* p) {
  return (unsigned)(unsigned long long)(__attribute__((address_space(3))) const
                                        void*)p;
}

// Async 16B HBM -> LDS copy, per-lane addresses, tracked by ASYNCcnt.
__device__ __forceinline__ void async_cp16(const _Float16* g, _Float16* l) {
  asm volatile("global_load_async_to_lds_b128 %0, %1, off" ::"v"(lds_off(l)),
               "v"((unsigned long long)g)
               : "memory");
}

// ---------------------------------------------------------------------------
// f32 -> f16 convert, 4 elements per thread
// ---------------------------------------------------------------------------
__global__ void cvt_f32_f16_kernel(const float* __restrict__ in,
                                   _Float16* __restrict__ out) {
  const int i = blockIdx.x * blockDim.x + threadIdx.x;
  const float4 f = ((const float4*)in)[i];
  v4h o;
  o[0] = (_Float16)f.x; o[1] = (_Float16)f.y;
  o[2] = (_Float16)f.z; o[3] = (_Float16)f.w;
  ((v4h*)out)[i] = o;
}

// ---------------------------------------------------------------------------
// f16 GEMM: 128x128 block tile, 256 threads (8 waves, 4(M) x 2(N)), each wave
// computes 32x64 = 2x4 WMMA tiles. K-step 32. A staged by async-to-LDS into a
// double buffer; B staged via registers with an LDS transpose so fragments are
// b128 LDS loads (As[m][k], Bs[n][k]).
// EPI: 0 = +bias(float) -> f16 out; 1 = f16 out; 2 = residual(f32)+acc -> f32
// ---------------------------------------------------------------------------
#define GLDA 40  // LDS row stride (f16 elems), 80B: 16B aligned, conflict-light

template <int EPI, int KC, int NC>
__global__ __launch_bounds__(256) void gemm_f16_kernel(
    const _Float16* __restrict__ A, const _Float16* __restrict__ B,
    const float* __restrict__ bias, const float* __restrict__ residual,
    _Float16* __restrict__ outH, float* __restrict__ outF) {
  __shared__ _Float16 As[2][128 * GLDA];
  __shared__ _Float16 Bs[128 * GLDA];

  const int tid = threadIdx.x;
  const int lane = tid & 31, w = tid >> 5;
  const int laneHi = (lane >> 4) & 1;
  const int ln16 = lane & 15;
  const int wm = w & 3, wn = w >> 2;  // 4 x 2 wave grid
  const int M0 = blockIdx.y * 128, N0 = blockIdx.x * 128;

  v8f acc[2][4];
#pragma unroll
  for (int i = 0; i < 2; i++)
#pragma unroll
    for (int j = 0; j < 4; j++) acc[i][j] = {};

  // A staging: thread -> (row, 16-elem half); two async b128 copies.
  const int arow = tid >> 1, ahalf = (tid & 1) * 16;
  // B staging: thread -> (n, 16-k half); 16 strided scalar loads (coalesced
  // across lanes), gathered into two aligned v8h LDS stores into Bs[n][k].
  const int bn = tid & 127, bkh = (tid >> 7) * 16;

  const _Float16* ga = A + (size_t)(M0 + arow) * KC + ahalf;
  const _Float16* gb = B + (size_t)bkh * NC + N0 + bn;

  auto async_a = [&](int k0, int buf) {
    _Float16* dst = &As[buf][arow * GLDA + ahalf];
    async_cp16(ga + k0, dst);
    async_cp16(ga + k0 + 8, dst + 8);
  };

  v8h b0, b1;
  auto stage_b = [&](int k0) {
    const _Float16* pb = gb + (size_t)k0 * NC;
#pragma unroll
    for (int i = 0; i < 8; i++) b0[i] = pb[(size_t)i * NC];
#pragma unroll
    for (int i = 0; i < 8; i++) b1[i] = pb[(size_t)(i + 8) * NC];
  };

  async_a(0, 0);
  stage_b(0);
  for (int k0 = 0; k0 < KC; k0 += 32) {
    const int p = (k0 >> 5) & 1;
    if (k0) __syncthreads();  // previous iteration's LDS readers complete
    *(v8h*)&Bs[bn * GLDA + bkh] = b0;
    *(v8h*)&Bs[bn * GLDA + bkh + 8] = b1;
    if (k0 + 32 < KC) {
      async_a(k0 + 32, p ^ 1);  // overlap next A tile with compute below
      stage_b(k0 + 32);
      asm volatile("s_wait_asynccnt 0x2" ::: "memory");  // A(cur) landed
    } else {
      asm volatile("s_wait_asynccnt 0x0" ::: "memory");
    }
    __syncthreads();

    v16h afr[2], bfr[4];
#pragma unroll
    for (int i = 0; i < 2; i++) {
      // A 16x32 fragment in this code's k order: 16 contiguous K starting at laneHi*16
      const _Float16* pa = &As[p][(wm * 32 + i * 16 + ln16) * GLDA];
      afr[i] = frag2(pa, laneHi * 16, laneHi * 16 + 8);
    }
#pragma unroll
    for (int j = 0; j < 4; j++) {
      // B 32x16 fragment in this code's k order (same as A): 16 contiguous K starting at laneHi*16
      const _Float16* pbf = &Bs[(wn * 64 + j * 16 + ln16) * GLDA];
      bfr[j] = frag2(pbf, laneHi * 16, laneHi * 16 + 8);
    }
#pragma unroll
    for (int i = 0; i < 2; i++)
#pragma unroll
      for (int j = 0; j < 4; j++) acc[i][j] = wmma16(afr[i], bfr[j], acc[i][j]);
  }

  // Epilogue (compile-time selected). C/D layout: elem e -> row = e + laneHi*8,
  // col = ln16.
#pragma unroll
  for (int i = 0; i < 2; i++) {
#pragma unroll
    for (int j = 0; j < 4; j++) {
      const int col = N0 + wn * 64 + j * 16 + ln16;
      const int row0 = M0 + wm * 32 + i * 16 + laneHi * 8;
      float bv = 0.0f;
      if (EPI == 0) bv = bias[col];
      const size_t idx0 = (size_t)row0 * NC + col;
#pragma unroll
      for (int e = 0; e < 8; e++) {
        const size_t idx = idx0 + (size_t)e * NC;
        const float v = acc[i][j][e];
        if (EPI == 0)      outH[idx] = (_Float16)(v + bv);
        else if (EPI == 1) outH[idx] = (_Float16)v;
        else               outF[idx] = residual[idx] + v;
      }
    }
  }
}

// ---------------------------------------------------------------------------
// h = layernorm(x) * (1 + scale) + shift  (row length 1024, ss row = [sc|sh])
// ---------------------------------------------------------------------------
__global__ __launch_bounds__(256) void mod_ln_kernel(
    const float* __restrict__ x, const _Float16* __restrict__ ss,
    _Float16* __restrict__ h) {
  const int r = blockIdx.x;
  const float* xr = x + (size_t)r * 1024;
  float v[4], lsum = 0.f, lsq = 0.f;
#pragma unroll
  for (int i = 0; i < 4; i++) {
    v[i] = xr[threadIdx.x + i * 256];
    lsum += v[i];
    lsq += v[i] * v[i];
  }
#pragma unroll
  for (int m = 16; m >= 1; m >>= 1) {
    lsum += __shfl_xor(lsum, m);
    lsq += __shfl_xor(lsq, m);
  }
  __shared__ float s1[8], s2[8];
  if ((threadIdx.x & 31) == 0) {
    s1[threadIdx.x >> 5] = lsum;
    s2[threadIdx.x >> 5] = lsq;
  }
  __syncthreads();
  float tsum = 0.f, tsq = 0.f;
#pragma unroll
  for (int i = 0; i < 8; i++) { tsum += s1[i]; tsq += s2[i]; }
  const float mu = tsum * (1.0f / 1024.0f);
  const float var = tsq * (1.0f / 1024.0f) - mu * mu;
  const float rstd = rsqrtf(var + 1e-5f);
  const _Float16* ssr = ss + (size_t)r * 2048;
#pragma unroll
  for (int i = 0; i < 4; i++) {
    const int c = threadIdx.x + i * 256;
    const float sc = (float)ssr[c];
    const float sh = (float)ssr[c + 1024];
    h[(size_t)r * 1024 + c] = (_Float16)(((v[i] - mu) * rstd) * (1.f + sc) + sh);
  }
}

// ---------------------------------------------------------------------------
// Per-head LN(D=64) + RoPE on q,k; repack q/k/v -> [B,H,N,D] f16.
// One wave per (b,n,h); lane l owns d=l and d=l+32 (the RoPE pair).
// ---------------------------------------------------------------------------
__global__ __launch_bounds__(256) void qk_ln_rope_kernel(
    const _Float16* __restrict__ qkv, _Float16* __restrict__ qP,
    _Float16* __restrict__ kP, _Float16* __restrict__ vP) {
  const int lane = threadIdx.x & 31;
  const int wid = blockIdx.x * 8 + (threadIdx.x >> 5);  // 0 .. B*N*H-1
  const int b = wid >> 15;            // / (N*H) = 32768
  const int rem = wid & 32767;
  const int n = rem >> 4, hh = rem & 15;

  const _Float16* qr = qkv + ((size_t)(b * 2048 + n)) * 3072 + hh * 64;
  float q1 = (float)qr[lane],        q2 = (float)qr[lane + 32];
  float k1 = (float)qr[1024 + lane], k2 = (float)qr[1024 + lane + 32];
  float v1 = (float)qr[2048 + lane], v2 = (float)qr[2048 + lane + 32];

#pragma unroll
  for (int t = 0; t < 2; t++) {
    float a = t ? k1 : q1, c = t ? k2 : q2;
    float s = a + c, sq = a * a + c * c;
#pragma unroll
    for (int m = 16; m >= 1; m >>= 1) {
      s += __shfl_xor(s, m);
      sq += __shfl_xor(sq, m);
    }
    const float mu = s * (1.0f / 64.0f);
    const float var = sq * (1.0f / 64.0f) - mu * mu;
    const float rstd = rsqrtf(var + 1e-5f);
    a = (a - mu) * rstd;
    c = (c - mu) * rstd;
    if (t) { k1 = a; k2 = c; } else { q1 = a; q2 = c; }
  }
  const float invf = __powf(10000.0f, -(float)lane * (1.0f / 32.0f));
  const float ang = (float)n * invf;
  float sn, cs;
  __sincosf(ang, &sn, &cs);

  const size_t o = ((size_t)((b * 16 + hh) * 2048 + n)) * 64;
  qP[o + lane]      = (_Float16)(q1 * cs - q2 * sn);
  qP[o + lane + 32] = (_Float16)(q1 * sn + q2 * cs);
  kP[o + lane]      = (_Float16)(k1 * cs - k2 * sn);
  kP[o + lane + 32] = (_Float16)(k1 * sn + k2 * cs);
  vP[o + lane]      = (_Float16)v1;
  vP[o + lane + 32] = (_Float16)v2;
}

// ---------------------------------------------------------------------------
// Flash attention. Block = 4 waves = 64 query rows for one (b,h).
// Loop over 32 key tiles of 64. Scores and P*V via WMMA; softmax fp32.
// K staged by async-to-LDS (double buffer); V register-staged + transposed.
// ---------------------------------------------------------------------------
#define ALDS 72  // 144B row stride: 16B aligned, conflict-light

__global__ __launch_bounds__(128) void attn_kernel(
    const _Float16* __restrict__ qP, const _Float16* __restrict__ kP,
    const _Float16* __restrict__ vP, float* __restrict__ oF,
    _Float16* __restrict__ oH) {
  __shared__ _Float16 ks[2][64 * ALDS];   // ks[key][d], double buffered
  __shared__ _Float16 vt[64 * ALDS];      // vt[d][key] (transposed)
  __shared__ _Float16 ps[4 * 16 * ALDS];  // per-wave P tiles [16][64]

  const int tid = threadIdx.x, lane = tid & 31, w = tid >> 5;
  const int laneHi = (lane >> 4) & 1, ln16 = lane & 15;
  const int bh = blockIdx.y;  // b*16 + h
  const int b = bh >> 4, hh = bh & 15;
  const size_t base = (size_t)bh * 2048 * 64;
  const int q0 = blockIdx.x * 64 + w * 16;

  // q fragments (A-layout, K=d, two 32-wide k-steps), loaded straight from HBM
  v16h qf[2];
  {
    const _Float16* qrow = qP + base + (size_t)(q0 + ln16) * 64;
    qf[0] = frag2(qrow, laneHi * 16, laneHi * 16 + 8);
    qf[1] = frag2(qrow + 32, laneHi * 16, laneHi * 16 + 8);
  }

  v8f oacc[4];
#pragma unroll
  for (int t = 0; t < 4; t++) oacc[t] = {};
  float mrow[8], lrow[8];
#pragma unroll
  for (int j = 0; j < 8; j++) { mrow[j] = -1e30f; lrow[j] = 0.f; }

  // staging: thread -> (key row, 32-elem half) of the 64x64 K and V tiles
  const int key = tid >> 1, half = (tid & 1) * 32;
  const _Float16* kg = kP + base + (size_t)key * 64 + half;
  const _Float16* vg = vP + base + (size_t)key * 64 + half;

  auto async_k = [&](int kt, int buf) {
    const _Float16* pk = kg + (size_t)kt * 4096;
    _Float16* dst = &ks[buf][key * ALDS + half];
#pragma unroll
    for (int u = 0; u < 4; u++) async_cp16(pk + u * 8, dst + u * 8);
  };

  v8h vr[4];
  auto stage_v = [&](int kt) {
    const _Float16* pv = vg + (size_t)kt * 4096;
#pragma unroll
    for (int u = 0; u < 4; u++) vr[u] = ((const v8h*)pv)[u];
  };

  async_k(0, 0);
  stage_v(0);
  for (int kt = 0; kt < 32; kt++) {
    const int p = kt & 1;
    if (kt) __syncthreads();  // prior readers of vt/ps/ks[p^1] done
#pragma unroll
    for (int u = 0; u < 4; u++)
#pragma unroll
      for (int e = 0; e < 8; e++)
        vt[(half + u * 8 + e) * ALDS + key] = vr[u][e];
    if (kt + 1 < 32) {
      async_k(kt + 1, p ^ 1);  // overlap next K tile with compute below
      stage_v(kt + 1);
      asm volatile("s_wait_asynccnt 0x4" ::: "memory");  // K(cur) landed
    } else {
      asm volatile("s_wait_asynccnt 0x0" ::: "memory");
    }
    __syncthreads();

    // scores s[m][key] = sum_d q*k ; B-frag cols = keys, contiguous d
    v8f sc[4];
#pragma unroll
    for (int t = 0; t < 4; t++) {
      const _Float16* kb = &ks[p][(t * 16 + ln16) * ALDS];
      v8f z = {};
      z = wmma16(qf[0], frag2(kb, laneHi * 16, laneHi * 16 + 8), z);
      z = wmma16(qf[1], frag2(kb + 32, laneHi * 16, laneHi * 16 + 8), z);
      sc[t] = z;
    }

    // online softmax; row j lives in 16-lane groups -> xor masks 8..1
#pragma unroll
    for (int j = 0; j < 8; j++) {
      float a0 = sc[0][j] * 0.125f, a1 = sc[1][j] * 0.125f;
      float a2 = sc[2][j] * 0.125f, a3 = sc[3][j] * 0.125f;
      float mx = fmaxf(fmaxf(a0, a1), fmaxf(a2, a3));
#pragma unroll
      for (int m = 8; m >= 1; m >>= 1) mx = fmaxf(mx, __shfl_xor(mx, m));
      const float mnew = fmaxf(mrow[j], mx);
      const float alpha = __expf(mrow[j] - mnew);
      const float p0 = __expf(a0 - mnew), p1 = __expf(a1 - mnew);
      const float p2 = __expf(a2 - mnew), p3 = __expf(a3 - mnew);
      float rs = p0 + p1 + p2 + p3;
#pragma unroll
      for (int m = 8; m >= 1; m >>= 1) rs += __shfl_xor(rs, m);
      lrow[j] = lrow[j] * alpha + rs;
      mrow[j] = mnew;
#pragma unroll
      for (int t = 0; t < 4; t++) oacc[t][j] *= alpha;
      _Float16* pr = &ps[(w * 16 + j + laneHi * 8) * ALDS];
      pr[ln16]      = (_Float16)p0;
      pr[16 + ln16] = (_Float16)p1;
      pr[32 + ln16] = (_Float16)p2;
      pr[48 + ln16] = (_Float16)p3;
    }
    __syncthreads();

    // O += P * V ; P as A-fragments from LDS, V^T B-fragments contiguous keys
    v16h pf[2];
#pragma unroll
    for (int kst = 0; kst < 2; kst++) {
      const _Float16* pr = &ps[(w * 16 + ln16) * ALDS + kst * 32];
      pf[kst] = frag2(pr, laneHi * 16, laneHi * 16 + 8);
    }
#pragma unroll
    for (int dt = 0; dt < 4; dt++) {
      const _Float16* vb = &vt[(dt * 16 + ln16) * ALDS];
#pragma unroll
      for (int kst = 0; kst < 2; kst++) {
        v16h vf = frag2(vb + kst * 32, laneHi * 16, laneHi * 16 + 8);
        oacc[dt] = wmma16(pf[kst], vf, oacc[dt]);
      }
    }
  }

  // normalize + write to [B,N,C] (f32 for residual, f16 for final GEMM A)
#pragma unroll
  for (int dt = 0; dt < 4; dt++) {
    const int d = dt * 16 + ln16;
#pragma unroll
    for (int j = 0; j < 8; j++) {
      const int qrow = q0 + j + laneHi * 8;
      const float val = oacc[dt][j] / lrow[j];
      const size_t idx = ((size_t)(b * 2048 + qrow)) * 1024 + hh * 64 + d;
      oF[idx] = val;
      oH[idx] = (_Float16)val;
    }
  }
}

// ---------------------------------------------------------------------------
extern "C" void kernel_launch(void* const* d_in, const int* in_sizes, int n_in,
                              void* d_out, int out_size, void* d_ws,
                              size_t ws_size, hipStream_t stream) {
  const float* x      = (const float*)d_in[0];
  const float* emb    = (const float*)d_in[1];
  const float* emb_w  = (const float*)d_in[2];
  const float* emb_b  = (const float*)d_in[3];
  const float* proj_w = (const float*)d_in[4];
  const float* out_w  = (const float*)d_in[5];
  float* out = (float*)d_out;

  char* ws = (char*)d_ws;
  const size_t MB = (size_t)1 << 20;
  _Float16* embH   = (_Float16*)(ws + 0 * MB);    // 16 MB  (8192x1024)
  _Float16* embwH  = (_Float16*)(ws + 16 * MB);   //  4 MB  (1024x2048)
  _Float16* projwH = (_Float16*)(ws + 20 * MB);   //  6 MB  (1024x3072)
  _Float16* outwH  = (_Float16*)(ws + 26 * MB);   //  2 MB  (1024x1024)
  _Float16* ssH    = (_Float16*)(ws + 28 * MB);   // 32 MB  (8192x2048)
  _Float16* hH     = (_Float16*)(ws + 60 * MB);   // 16 MB  (8192x1024)
  _Float16* qkvH   = (_Float16*)(ws + 76 * MB);   // 48 MB  (8192x3072)
  _Float16* qPack  = (_Float16*)(ws + 124 * MB);  // 16 MB  [B,H,N,D]
  _Float16* kPack  = (_Float16*)(ws + 140 * MB);  // 16 MB
  _Float16* vPack  = (_Float16*)(ws + 156 * MB);  // 16 MB
  _Float16* oH     = (_Float16*)(ws + 172 * MB);  // 16 MB  (8192x1024)
  float*    oF     = (float*)   (ws + 188 * MB);  // 32 MB  (8192x1024)

  // 1) converts (4 elems/thread)
  cvt_f32_f16_kernel<<<8388608 / 1024, 256, 0, stream>>>(emb, embH);
  cvt_f32_f16_kernel<<<2097152 / 1024, 256, 0, stream>>>(emb_w, embwH);
  cvt_f32_f16_kernel<<<3145728 / 1024, 256, 0, stream>>>(proj_w, projwH);
  cvt_f32_f16_kernel<<<1048576 / 1024, 256, 0, stream>>>(out_w, outwH);

  // 2) ss = emb @ emb_w + emb_b   (M=8192, K=1024, N=2048)
  gemm_f16_kernel<0, 1024, 2048><<<dim3(2048 / 128, 8192 / 128), 256, 0,
                                   stream>>>(embH, embwH, emb_b, nullptr, ssH,
                                             nullptr);

  // 3) h = LN(x)*(1+scale)+shift
  mod_ln_kernel<<<8192, 256, 0, stream>>>(x, ssH, hH);

  // 4) qkv = h @ proj_w           (M=8192, K=1024, N=3072)
  gemm_f16_kernel<1, 1024, 3072><<<dim3(3072 / 128, 8192 / 128), 256, 0,
                                   stream>>>(hH, projwH, nullptr, nullptr, qkvH,
                                             nullptr);

  // 5) per-head LN + RoPE + repack (B*N*H waves, 8 per block)
  qk_ln_rope_kernel<<<131072 / 8, 256, 0, stream>>>(qkvH, qPack, kPack, vPack);

  // 6) flash attention: 32 query tiles x 64 (b,h) pairs
  attn_kernel<<<dim3(32, 64), 128, 0, stream>>>(qPack, kPack, vPack, oF, oH);

  // 7) out = o + o @ out_w        (M=8192, K=1024, N=1024)
  gemm_f16_kernel<2, 1024, 1024><<<dim3(1024 / 128, 8192 / 128), 256, 0,
                                   stream>>>(oH, outwH, nullptr, oF, nullptr,
                                             out);
}
